// PerformerHead_18683107737673
// MI455X (gfx1250) — compile-verified, matrix operand order checked by reading, not run on hardware
//
#include <hip/hip_runtime.h>
#include <hip/hip_bf16.h>

// ---------------------------------------------------------------------------
// Performer (FAVOR+) forward for MI455X / gfx1250.
// All matmuls via v_wmma_f32_16x16x32_f16 (f16 in, f32 accumulate).
// Attention chunk staging (qf/kf tiles) via Tensor Data Mover
// (tensor_load_to_lds + s_wait_tensorcnt), with TDM LDS padding used to
// produce the bank-conflict-avoiding 136-half row stride.
// ---------------------------------------------------------------------------

typedef _Float16 f16;
typedef _Float16 h2   __attribute__((ext_vector_type(2)));
typedef _Float16 h8   __attribute__((ext_vector_type(8)));
typedef _Float16 v16h __attribute__((ext_vector_type(16)));
typedef float    v8f  __attribute__((ext_vector_type(8)));
typedef float    f4   __attribute__((ext_vector_type(4)));
typedef unsigned int u32x4 __attribute__((ext_vector_type(4)));
typedef int          i32x4 __attribute__((ext_vector_type(4)));
typedef int          i32x8 __attribute__((ext_vector_type(8)));

#define HEADS   4
#define DHEAD   64
#define DMODEL  256
#define MFEAT   128
#define SEQ     4096
#define BATCH   16
#define ROWS    (BATCH * SEQ)      // 65536
#define CHUNKN  128
#define NCHUNK  (SEQ / CHUNKN)     // 32
#define LAYERS  4

__device__ __forceinline__ v8f wmma32(v16h a, v16h b, v8f c) {
  return __builtin_amdgcn_wmma_f32_16x16x32_f16(false, a, false, b, (short)0, c,
                                                false, false);
}

// Fragment load from LDS laid out [row][k] (row stride ldk halves).
// Used for both A (row = M index) and B (row = N index) operands; element i
// maps to k = pair(v) + (i&1) + 8*(lane>=16), matching the ISA 16-bit layout.
__device__ __forceinline__ v16h load_frag(const f16* base, int row, int kbase,
                                          int ldk, int lane) {
  int r = row + (lane & 15);
  int hf = lane >> 4;
  const f16* p = base + r * ldk + kbase + hf * 8;
  v16h f;
#pragma unroll
  for (int v = 0; v < 8; ++v) {
    int k = (v < 4) ? (v * 2) : (16 + (v - 4) * 2);
    h2 pr = *(const h2*)(p + k);
    f[2 * v]     = pr[0];
    f[2 * v + 1] = pr[1];
  }
  return f;
}

// Logical A[row][k] where storage is base[k][row] (transposed read).
__device__ __forceinline__ v16h load_frag_T(const f16* base, int row, int kbase,
                                            int ldk, int lane) {
  int r = row + (lane & 15);
  int hf = lane >> 4;
  v16h f;
#pragma unroll
  for (int i = 0; i < 16; ++i) {
    int v = i >> 1;
    int k = ((v < 4) ? (v * 2) : (16 + (v - 4) * 2)) + (i & 1) + hf * 8 + kbase;
    f[i] = base[k * ldk + r];
  }
  return f;
}

__device__ __forceinline__ float gelu_f(float x) {
  float x3 = x * x * x;
  return 0.5f * x * (1.0f + tanhf(0.7978845608028654f * (x + 0.044715f * x3)));
}

// ---------------------------------------------------------------------------
// Tensor Data Mover: async-copy a 2D f16 tile (tile_d1 rows x tile_d0 elems,
// global row stride stride_elems) into LDS at lds_byte_addr. When pad_en,
// pad_interval/pad_amount (encoded per D# group1) insert LDS padding per row.
// D# layout per CDNA5 ISA.
// ---------------------------------------------------------------------------
__device__ __forceinline__ void tdm_load_tile_f16(
    unsigned lds_byte_addr, const f16* gptr, unsigned tensor_d0,
    unsigned tensor_d1, unsigned tile_d0, unsigned tile_d1,
    unsigned stride_elems, unsigned pad_interval_code,
    unsigned pad_amount_code, bool pad_en) {
  unsigned long long ga = (unsigned long long)(uintptr_t)gptr;
  u32x4 g0;
  g0[0] = 1u;                                   // count=1, user descriptor
  g0[1] = lds_byte_addr;                        // lds_addr
  g0[2] = (unsigned)(ga & 0xFFFFFFFFull);       // global_addr[31:0]
  g0[3] = (unsigned)((ga >> 32) & 0x01FFFFFFull) | (2u << 30);  // type=2
  i32x8 g1;
  g1[0] = (int)((1u << 16) |                    // data_size = 1 -> 2 bytes
                (pad_en ? (1u << 20) : 0u) |
                (pad_interval_code << 22) | (pad_amount_code << 25));
  g1[1] = (int)(tensor_d0 << 16);               // tensor_dim0 lo16
  g1[2] = (int)(((tensor_d0 >> 16) & 0xFFFFu) | (tensor_d1 << 16));
  g1[3] = (int)(((tensor_d1 >> 16) & 0xFFFFu) | (tile_d0 << 16));
  g1[4] = (int)(tile_d1 & 0xFFFFu);             // tile_dim1 (tile_dim2 = 0)
  g1[5] = (int)stride_elems;                    // tensor_dim0_stride lo32
  g1[6] = 0;
  g1[7] = 0;
  i32x4 z4 = {0, 0, 0, 0};
#if defined(__clang_major__) && (__clang_major__ >= 23)
  i32x8 z8 = {0, 0, 0, 0, 0, 0, 0, 0};
  __builtin_amdgcn_tensor_load_to_lds(g0, g1, z4, z4, z8, 0);
#else
  __builtin_amdgcn_tensor_load_to_lds(g0, g1, z4, z4, 0);
#endif
}

// ---------------------------------------------------------------------------
// Embedding: h[b,n,:] = emb_tok[x[b,n]] + emb_pos[n]
// ---------------------------------------------------------------------------
__global__ __launch_bounds__(256) void embed_kernel(
    const int* __restrict__ x, const float* __restrict__ etok,
    const float* __restrict__ epos, float* __restrict__ h) {
  int g = blockIdx.x * 256 + threadIdx.x;  // over ROWS*DMODEL
  int d = g & (DMODEL - 1);
  int row = g >> 8;
  int n = row & (SEQ - 1);
  int tok = x[row];
  h[g] = etok[tok * DMODEL + d] + epos[n * DMODEL + d];
}

// ---------------------------------------------------------------------------
// LayerNorm over D=256, one token per block, f16 output for WMMA consumption.
// ---------------------------------------------------------------------------
__global__ __launch_bounds__(256) void layernorm_kernel(
    const float* __restrict__ h, const float* __restrict__ s,
    const float* __restrict__ b, f16* __restrict__ out) {
  __shared__ float r1[256];
  __shared__ float r2[256];
  int row = blockIdx.x, t = threadIdx.x;
  float v = h[(long)row * DMODEL + t];
  r1[t] = v;
  r2[t] = v * v;
  __syncthreads();
  for (int o = 128; o > 0; o >>= 1) {
    if (t < o) { r1[t] += r1[t + o]; r2[t] += r2[t + o]; }
    __syncthreads();
  }
  float mu  = r1[0] * (1.0f / 256.0f);
  float var = r2[0] * (1.0f / 256.0f) - mu * mu;
  float inv = rsqrtf(var + 1e-5f);
  out[(long)row * DMODEL + t] = (f16)((v - mu) * inv * s[t] + b[t]);
}

// ---------------------------------------------------------------------------
// Generic WMMA GEMM: C[M x N] = act(A_f16 @ B_f32 + bias) (+ residual)
// Block tile 128x128, 8 waves, each wave 32x64 (2x4 fragments), K step 32.
// ---------------------------------------------------------------------------
template <bool TRANSB, bool GELU, bool RES, bool OUT16, bool OUT32, bool BIAS>
__global__ __launch_bounds__(256) void gemm_f16_kernel(
    const f16* __restrict__ A, int lda,
    const float* __restrict__ B, int ldb,
    const float* __restrict__ bias,
    const float* __restrict__ res,
    float* __restrict__ C32, f16* __restrict__ C16,
    int N, int K) {
  __shared__ f16 As[128][40];  // [m][k]
  __shared__ f16 Bs[128][40];  // [n][k]
  int tid = threadIdx.x;
  int lane = tid & 31, w = tid >> 5;
  int mtile = blockIdx.y * 128, ntile = blockIdx.x * 128;
  int wm = (w >> 1) * 32, wn = (w & 1) * 64;

  v8f acc[2][4] = {};
  int arow = tid >> 1, aseg = (tid & 1) * 16;

  for (int k0 = 0; k0 < K; k0 += 32) {
    const f16* ap = A + (long)(mtile + arow) * lda + k0 + aseg;
    *(h8*)&As[arow][aseg]     = *(const h8*)(ap);
    *(h8*)&As[arow][aseg + 8] = *(const h8*)(ap + 8);
    if (k0 + 32 < K) __builtin_prefetch(ap + 32, 0, 0);
    if (TRANSB) {
      const float* bp = B + (long)(ntile + arow) * ldb + k0 + aseg;
#pragma unroll
      for (int i = 0; i < 16; i += 4) {
        f4 t4 = *(const f4*)(bp + i);
#pragma unroll
        for (int q = 0; q < 4; ++q) Bs[arow][aseg + i + q] = (f16)t4[q];
      }
    } else {
#pragma unroll
      for (int i = 0; i < 16; ++i)
        Bs[arow][aseg + i] = (f16)B[(long)(k0 + aseg + i) * ldb + ntile + arow];
    }
    __syncthreads();
    v16h af[2], bf[4];
#pragma unroll
    for (int rb = 0; rb < 2; ++rb)
      af[rb] = load_frag(&As[0][0], wm + rb * 16, 0, 40, lane);
#pragma unroll
    for (int cb = 0; cb < 4; ++cb)
      bf[cb] = load_frag(&Bs[0][0], wn + cb * 16, 0, 40, lane);
#pragma unroll
    for (int rb = 0; rb < 2; ++rb)
#pragma unroll
      for (int cb = 0; cb < 4; ++cb)
        acc[rb][cb] = wmma32(af[rb], bf[cb], acc[rb][cb]);
    __syncthreads();
  }

  int colL = lane & 15, hf = lane >> 4;
#pragma unroll
  for (int cb = 0; cb < 4; ++cb) {
    int n = ntile + wn + cb * 16 + colL;
    float bv = BIAS ? bias[n] : 0.0f;
#pragma unroll
    for (int rb = 0; rb < 2; ++rb) {
#pragma unroll
      for (int j = 0; j < 8; ++j) {
        int m = mtile + wm + rb * 16 + j + hf * 8;
        float vv = acc[rb][cb][j] + bv;
        if (GELU) vv = gelu_f(vv);
        if (RES) vv += res[(long)m * N + n];
        if (OUT32) C32[(long)m * N + n] = vv;
        if (OUT16) C16[(long)m * N + n] = (f16)vv;
      }
    }
  }
}

// ---------------------------------------------------------------------------
// FAVOR+ features. MODE 0: query (fused row-max). MODE 1: key pass-1 (block
// max only). MODE 2: key pass-2 (apply global per-(b,h) max).
// xp = dn * (x @ proj^T), out = ratio*(exp(xp - diag - mx) + eps)
// ---------------------------------------------------------------------------
template <int MODE>
__global__ __launch_bounds__(256) void favor_kernel(
    const f16* __restrict__ X,       // (ROWS, 256) f16, head slice via +h*64
    const float* __restrict__ proj,  // (128, 64)
    f16* __restrict__ OUT,           // (B,H,N,M) f16
    float* __restrict__ bmax,        // [H*512]
    const float* __restrict__ gmax)  // [B*H]
{
  extern __shared__ char smem[];
  f16(*As)[40]    = (f16(*)[40])smem;                   // 10240 B
  f16(*Bs)[40]    = (f16(*)[40])(smem + 10240);         // 10240 B
  float(*xps)[132] = (float(*)[132])(smem + 20480);     // 67584 B
  float* diag = (float*)(smem + 20480 + 67584);         // 512 B
  float* rmax = diag + 128;                             // 512 B

  int tid = threadIdx.x, lane = tid & 31, w = tid >> 5;
  int mtile = blockIdx.x * 128;
  int head  = blockIdx.y;
  int wm = (w >> 1) * 32, wn = (w & 1) * 64;
  v8f acc[2][4] = {};
  int arow = tid >> 1, aseg = (tid & 1) * 16;

  for (int k0 = 0; k0 < DHEAD; k0 += 32) {
    const f16* ap = X + (long)(mtile + arow) * DMODEL + head * DHEAD + k0 + aseg;
    *(h8*)&As[arow][aseg]     = *(const h8*)(ap);
    *(h8*)&As[arow][aseg + 8] = *(const h8*)(ap + 8);
    const float* bp = proj + arow * DHEAD + k0 + aseg;  // B^T: [m][dh]
#pragma unroll
    for (int i = 0; i < 16; i += 4) {
      f4 t4 = *(const f4*)(bp + i);
#pragma unroll
      for (int q = 0; q < 4; ++q) Bs[arow][aseg + i + q] = (f16)t4[q];
    }
    __syncthreads();
    v16h af[2], bf[4];
#pragma unroll
    for (int rb = 0; rb < 2; ++rb)
      af[rb] = load_frag(&As[0][0], wm + rb * 16, 0, 40, lane);
#pragma unroll
    for (int cb = 0; cb < 4; ++cb)
      bf[cb] = load_frag(&Bs[0][0], wn + cb * 16, 0, 40, lane);
#pragma unroll
    for (int rb = 0; rb < 2; ++rb)
#pragma unroll
      for (int cb = 0; cb < 4; ++cb)
        acc[rb][cb] = wmma32(af[rb], bf[cb], acc[rb][cb]);
    __syncthreads();
  }

  const float DN = 0.35355339059327373f;  // 64^-0.25
  int colL = lane & 15, hf = lane >> 4;
#pragma unroll
  for (int cb = 0; cb < 4; ++cb)
#pragma unroll
    for (int rb = 0; rb < 2; ++rb)
#pragma unroll
      for (int j = 0; j < 8; ++j)
        xps[wm + rb * 16 + j + hf * 8][wn + cb * 16 + colL] =
            acc[rb][cb][j] * DN;
  __syncthreads();

  if (MODE != 1) {
    if (tid < 128) {
      float ss = 0.0f;
      const f16* xp = X + (long)(mtile + tid) * DMODEL + head * DHEAD;
      for (int d = 0; d < DHEAD; ++d) {
        float xv = (float)xp[d];
        ss += xv * xv;
      }
      diag[tid] = ss * 0.0625f;  // 0.5 * dn^2 = 0.5 * 64^-0.5
    }
  }
  if (MODE == 0 || MODE == 1) {
    if (tid < 128) {
      float m = -1e30f;
      for (int c = 0; c < 128; ++c) m = fmaxf(m, xps[tid][c]);
      rmax[tid] = m;
    }
  }
  __syncthreads();

  if (MODE == 1) {
    if (tid == 0) {
      float m = -1e30f;
      for (int r = 0; r < 128; ++r) m = fmaxf(m, rmax[r]);
      bmax[head * 512 + blockIdx.x] = m;
    }
    return;
  }

  float gm = 0.0f;
  if (MODE == 2) gm = gmax[(mtile >> 12) * HEADS + head];
  const float RATIO = 0.08838834764831845f;  // 128^-0.5
  for (int e = tid; e < 128 * 128; e += 256) {
    int r = e >> 7, c = e & 127;
    float mx = (MODE == 0) ? rmax[r] : gm;
    float val = RATIO * (expf(xps[r][c] - diag[r] - mx) + 1e-4f);
    int grow = mtile + r;
    int bb = grow >> 12, nn = grow & (SEQ - 1);
    OUT[((long)(bb * HEADS + head) * SEQ + nn) * MFEAT + c] = (f16)val;
  }
}

__global__ void kmax_reduce_kernel(const float* __restrict__ bmax,
                                   float* __restrict__ gmax) {
  int t = threadIdx.x;  // 64 = B*H
  if (t < BATCH * HEADS) {
    int b = t >> 2, h = t & 3;
    float m = -1e30f;
    for (int rb = 0; rb < 32; ++rb) m = fmaxf(m, bmax[h * 512 + b * 32 + rb]);
    gmax[t] = m;
  }
}

// ---------------------------------------------------------------------------
// Causal linear attention. One workgroup per (b,h); S(128x64 f32) and z(128)
// stay in LDS across the 32-chunk scan. All chunk matmuls are WMMA. qf/kf
// chunk tiles staged by the Tensor Data Mover (async, TENSORcnt-tracked) with
// hardware LDS padding giving the 136-half row stride.
// ---------------------------------------------------------------------------
#define QH_OFF (32768 + 1024)        // byte offset of qh in dynamic LDS
#define KH_OFF (QH_OFF + 34816)      // byte offset of kh

__global__ __launch_bounds__(256) void attention_kernel(
    const f16* __restrict__ QF, const f16* __restrict__ KF,
    const f16* __restrict__ V, f16* __restrict__ O) {
  extern __shared__ char smem[];
  float* S32 = (float*)smem;                 // 128*64 f32       = 32768 B
  float* z   = (float*)(smem + 32768);       // 128 f32
  float* den = z + 128;                      // 128 f32
  f16* qh = (f16*)(smem + QH_OFF);           // [i][m] 128x136   = 34816 B
  f16* kh = (f16*)(smem + KH_OFF);           // [j][m] 128x136   = 34816 B
  f16* ah = kh + 128 * 136;                  // [i][j] 128x136   = 34816 B
  f16* vh = ah + 128 * 136;                  // [d][j]  64x136   = 17408 B
  f16* Sh = vh + 64 * 136;                   // [d][m]  64x136   = 17408 B

  int tid = threadIdx.x, lane = tid & 31, w = tid >> 5;
  int bh = blockIdx.x;  // b*HEADS + h
  int b = bh >> 2, h = bh & 3;
  const f16* qbase = QF + (long)bh * SEQ * MFEAT;
  const f16* kbase = KF + (long)bh * SEQ * MFEAT;
  int colL = lane & 15, hf = lane >> 4;
  unsigned ldsbase = __builtin_amdgcn_groupstaticsize();

  for (int e = tid; e < 128 * 64; e += 256) S32[e] = 0.0f;
  if (tid < 128) z[tid] = 0.0f;
  __syncthreads();

  for (int c = 0; c < NCHUNK; ++c) {
    // ---- Phase A: TDM-stage qf/kf tiles; stage v chunk + f16 copy of S ----
    if (tid < 32) {  // wave 0 issues the tensor DMAs
      // tile row = 128 halves = 64 DWORDs -> pad_interval code 5;
      // pad 8 halves = 4 DWORDs -> pad_amount code 3 => LDS stride 136 halves.
      tdm_load_tile_f16(ldsbase + QH_OFF, qbase + (long)c * CHUNKN * MFEAT,
                        MFEAT, SEQ - c * CHUNKN, MFEAT, CHUNKN, MFEAT, 5, 3,
                        true);
      tdm_load_tile_f16(ldsbase + KH_OFF, kbase + (long)c * CHUNKN * MFEAT,
                        MFEAT, SEQ - c * CHUNKN, MFEAT, CHUNKN, MFEAT, 5, 3,
                        true);
      __builtin_amdgcn_s_wait_tensorcnt(0);
    }
    for (int e = tid; e < 128 * 64; e += 256) {
      int j = e >> 6, d = e & 63;
      vh[d * 136 + j] = V[(long)(b * SEQ + c * CHUNKN + j) * DMODEL + h * DHEAD + d];
    }
    for (int e = tid; e < 128 * 64; e += 256) {
      int m = e >> 6, d = e & 63;
      Sh[d * 136 + m] = (f16)S32[m * 64 + d];
    }
    __syncthreads();

    // ---- Phase B: attn = qf @ kf^T (128x128, K=128), masked, -> ah f16 ----
    {
      int wm = (w >> 1) * 32, wn = (w & 1) * 64;
      v8f acc[2][4] = {};
      for (int k0 = 0; k0 < 128; k0 += 32) {
        v16h af[2], bf[4];
#pragma unroll
        for (int rb = 0; rb < 2; ++rb)
          af[rb] = load_frag(qh, wm + rb * 16, k0, 136, lane);
#pragma unroll
        for (int cb = 0; cb < 4; ++cb)
          bf[cb] = load_frag(kh, wn + cb * 16, k0, 136, lane);
#pragma unroll
        for (int rb = 0; rb < 2; ++rb)
#pragma unroll
          for (int cb = 0; cb < 4; ++cb)
            acc[rb][cb] = wmma32(af[rb], bf[cb], acc[rb][cb]);
      }
#pragma unroll
      for (int cb = 0; cb < 4; ++cb)
#pragma unroll
        for (int rb = 0; rb < 2; ++rb)
#pragma unroll
          for (int j = 0; j < 8; ++j) {
            int i  = wm + rb * 16 + j + hf * 8;
            int jj = wn + cb * 16 + colL;
            float vv = (jj <= i) ? acc[rb][cb][j] : 0.0f;  // causal tril mask
            ah[i * 136 + jj] = (f16)vv;
          }
    }
    __syncthreads();

    // ---- Phase C: den[i] = rowsum(attn) + qf[i] . z  (pre-update z) ----
    if (tid < 128) {
      float s = 0.0f;
      for (int j = 0; j < 128; ++j) s += (float)ah[tid * 136 + j];
      for (int m = 0; m < 128; ++m) s += (float)qh[tid * 136 + m] * z[m];
      den[tid] = s;
    }
    __syncthreads();

    // ---- Phase D: num = attn @ v + qf @ S ; out = num / den ----
    {
      int wm2 = w * 16;
      v8f acc2[4] = {};
      for (int k0 = 0; k0 < 128; k0 += 32) {
        v16h af = load_frag(ah, wm2, k0, 136, lane);
#pragma unroll
        for (int cb = 0; cb < 4; ++cb) {
          v16h bf = load_frag(vh, cb * 16, k0, 136, lane);
          acc2[cb] = wmma32(af, bf, acc2[cb]);
        }
      }
      for (int k0 = 0; k0 < 128; k0 += 32) {
        v16h af = load_frag(qh, wm2, k0, 136, lane);
#pragma unroll
        for (int cb = 0; cb < 4; ++cb) {
          v16h bf = load_frag(Sh, cb * 16, k0, 136, lane);
          acc2[cb] = wmma32(af, bf, acc2[cb]);
        }
      }
#pragma unroll
      for (int cb = 0; cb < 4; ++cb)
#pragma unroll
        for (int j = 0; j < 8; ++j) {
          int i = wm2 + j + hf * 8;
          int d = cb * 16 + colL;
          float vv = acc2[cb][j] / den[i];
          O[(long)(b * SEQ + c * CHUNKN + i) * DMODEL + h * DHEAD + d] = (f16)vv;
        }
    }

    // ---- Phase E: S += kf^T @ v (rows m distributed across waves) ----
    {
      int wm3 = w * 16;
      v8f acc3[4] = {};
      for (int k0 = 0; k0 < 128; k0 += 32) {
        v16h af = load_frag_T(kh, wm3, k0, 136, lane);  // A[m][j] = kh[j][m]
#pragma unroll
        for (int cb = 0; cb < 4; ++cb) {
          v16h bf = load_frag(vh, cb * 16, k0, 136, lane);
          acc3[cb] = wmma32(af, bf, acc3[cb]);
        }
      }
#pragma unroll
      for (int cb = 0; cb < 4; ++cb)
#pragma unroll
        for (int j = 0; j < 8; ++j) {
          int m = wm3 + j + hf * 8;
          int d = cb * 16 + colL;
          S32[m * 64 + d] += acc3[cb][j];
        }
    }

    // ---- Phase F: z[m] += sum_j kf[j][m] ----
    if (tid < 128) {
      float s = 0.0f;
      for (int j = 0; j < 128; ++j) s += (float)kh[j * 136 + tid];
      z[tid] += s;
    }
    __syncthreads();
  }
}

// ---------------------------------------------------------------------------
// Classifier: out[b, c] = h[b, 0, :] @ w_cls + b_cls (tiny; scalar)
// ---------------------------------------------------------------------------
__global__ void classifier_kernel(const float* __restrict__ h,
                                  const float* __restrict__ wc,
                                  const float* __restrict__ bc,
                                  float* __restrict__ out) {
  int t = threadIdx.x;
  if (t < BATCH * 10) {
    int b = t / 10, c = t % 10;
    const float* hp = h + (long)b * SEQ * DMODEL;  // token 0
    float s = bc[c];
    for (int d = 0; d < DMODEL; ++d) s += hp[d] * wc[d * 10 + c];
    out[b * 10 + c] = s;
  }
}

// ---------------------------------------------------------------------------
extern "C" void kernel_launch(void* const* d_in, const int* in_sizes, int n_in,
                              void* d_out, int out_size, void* d_ws,
                              size_t ws_size, hipStream_t stream) {
  (void)in_sizes; (void)n_in; (void)out_size; (void)ws_size;
  const int*   x    = (const int*)d_in[0];
  const float* etok = (const float*)d_in[1];
  const float* epos = (const float*)d_in[2];
  const float* proj = (const float*)d_in[3];
  const float* ln1s = (const float*)d_in[4];
  const float* ln1b = (const float*)d_in[5];
  const float* wq   = (const float*)d_in[6];
  const float* wk   = (const float*)d_in[7];
  const float* wv   = (const float*)d_in[8];
  const float* wo   = (const float*)d_in[9];
  const float* bo   = (const float*)d_in[10];
  const float* ln2s = (const float*)d_in[11];
  const float* ln2b = (const float*)d_in[12];
  const float* w1   = (const float*)d_in[13];
  const float* b1   = (const float*)d_in[14];
  const float* w2   = (const float*)d_in[15];
  const float* b2   = (const float*)d_in[16];
  const float* wcls = (const float*)d_in[17];
  const float* bcls = (const float*)d_in[18];

  // Workspace layout (bytes). h1 reuses the q/k/v/qf region (free post-attn).
  char* ws = (char*)d_ws;
  float* h    = (float*)(ws + 0);            // 64 MB
  f16* lnbuf  = (f16*)(ws + 67108864);       // 32 MB
  f16* q16    = (f16*)(ws + 100663296);      // 32 MB
  f16* k16    = (f16*)(ws + 134217728);      // 32 MB
  f16* v16    = (f16*)(ws + 167772160);      // 32 MB
  f16* qf     = (f16*)(ws + 201326592);      // 64 MB
  f16* kf     = (f16*)(ws + 268435456);      // 64 MB
  f16* aout   = (f16*)(ws + 335544320);      // 32 MB
  float* bmax = (float*)(ws + 369098752);    // 8 KB
  float* gmax = (float*)(ws + 369107200);    // 256 B
  f16* h1     = (f16*)(ws + 100663296);      // 128 MB (reuse)

  const size_t favShm = 20480 + 67584 + 1024;      // 89088 B
  const size_t attShm = 32768 + 1024 + 3 * 34816 + 2 * 17408;  // 173056 B
  (void)hipFuncSetAttribute((const void*)&favor_kernel<0>,
      hipFuncAttributeMaxDynamicSharedMemorySize, (int)favShm);
  (void)hipFuncSetAttribute((const void*)&favor_kernel<1>,
      hipFuncAttributeMaxDynamicSharedMemorySize, (int)favShm);
  (void)hipFuncSetAttribute((const void*)&favor_kernel<2>,
      hipFuncAttributeMaxDynamicSharedMemorySize, (int)favShm);
  (void)hipFuncSetAttribute((const void*)&attention_kernel,
      hipFuncAttributeMaxDynamicSharedMemorySize, (int)attShm);

  dim3 blk(256);
  embed_kernel<<<ROWS * DMODEL / 256, blk, 0, stream>>>(x, etok, epos, h);

  for (int l = 0; l < LAYERS; ++l) {
    const float* projl = proj + (size_t)l * MFEAT * DHEAD;
    layernorm_kernel<<<ROWS, blk, 0, stream>>>(h, ln1s + l * 256, ln1b + l * 256, lnbuf);

    // QKV projections (f16 WMMA)
    gemm_f16_kernel<false, false, false, true, false, false>
        <<<dim3(2, 512), blk, 0, stream>>>(lnbuf, 256, wq + (size_t)l * 65536, 256,
                                           nullptr, nullptr, nullptr, q16, 256, 256);
    gemm_f16_kernel<false, false, false, true, false, false>
        <<<dim3(2, 512), blk, 0, stream>>>(lnbuf, 256, wk + (size_t)l * 65536, 256,
                                           nullptr, nullptr, nullptr, k16, 256, 256);
    gemm_f16_kernel<false, false, false, true, false, false>
        <<<dim3(2, 512), blk, 0, stream>>>(lnbuf, 256, wv + (size_t)l * 65536, 256,
                                           nullptr, nullptr, nullptr, v16, 256, 256);

    // FAVOR+ features
    favor_kernel<0><<<dim3(512, 4), blk, favShm, stream>>>(q16, projl, qf, nullptr, nullptr);
    favor_kernel<1><<<dim3(512, 4), blk, favShm, stream>>>(k16, projl, nullptr, bmax, nullptr);
    kmax_reduce_kernel<<<1, 64, 0, stream>>>(bmax, gmax);
    favor_kernel<2><<<dim3(512, 4), blk, favShm, stream>>>(k16, projl, kf, nullptr, gmax);

    // Causal linear attention scan (S, z resident in LDS; TDM staging)
    attention_kernel<<<BATCH * HEADS, blk, attShm, stream>>>(qf, kf, v16, aout);

    // Output projection + bias + residual into h (fp32)
    gemm_f16_kernel<false, false, true, false, true, true>
        <<<dim3(2, 512), blk, 0, stream>>>(aout, 256, wo + (size_t)l * 65536, 256,
                                           bo + l * 256, h, h, nullptr, 256, 256);

    // MLP
    layernorm_kernel<<<ROWS, blk, 0, stream>>>(h, ln2s + l * 256, ln2b + l * 256, lnbuf);
    gemm_f16_kernel<false, true, false, true, false, true>
        <<<dim3(8, 512), blk, 0, stream>>>(lnbuf, 256, w1 + (size_t)l * 262144, 1024,
                                           b1 + l * 1024, nullptr, nullptr, h1, 1024, 256);
    gemm_f16_kernel<false, false, true, false, true, true>
        <<<dim3(2, 512), blk, 0, stream>>>(h1, 1024, w2 + (size_t)l * 262144, 256,
                                           b2 + l * 256, h, h, nullptr, 256, 1024);
  }

  classifier_kernel<<<1, 256, 0, stream>>>(h, wcls, bcls, (float*)d_out);
}
